// RecurrentEncoder_5446018532087
// MI455X (gfx1250) — matrix operand order repaired and checked by reading, not compiled, not run on hardware
//
#include <hip/hip_runtime.h>

typedef __attribute__((ext_vector_type(8)))  __bf16 v8bf;
typedef __attribute__((ext_vector_type(16))) __bf16 v16bf;
typedef __attribute__((ext_vector_type(8)))  float  v8f;
typedef int v4i_ __attribute__((vector_size(16)));   // matches builtin param type

#define BLK_M 128
#define BLK_N 128
#define BLK_K 32
#define LDSS  40   // 32 + 8 pad elements; keeps 16B alignment, spreads banks

#if __has_builtin(__builtin_amdgcn_global_load_async_to_lds_b128)
#define HAVE_ASYNC_LDS 1
#else
#define HAVE_ASYNC_LDS 0
#endif

#define AS1 __attribute__((address_space(1)))
#define AS3 __attribute__((address_space(3)))

__device__ __forceinline__ void async_copy_b128(const __bf16* gsrc, __bf16* ldst) {
#if HAVE_ASYNC_LDS
    // (global_src AS1 int4*, lds_dst AS3 int4*, imm offset, imm cpol)
    // integer-cast route: AS1 ptrs are 64-bit; AS3 ptrs are 32-bit LDS offsets
    __builtin_amdgcn_global_load_async_to_lds_b128(
        (AS1 v4i_*)(uintptr_t)gsrc,
        (AS3 v4i_*)(unsigned)(uintptr_t)ldst, 0, 0);
#else
    (void)gsrc; (void)ldst;
#endif
}

template <int N>
__device__ __forceinline__ void wait_asynccnt() {
#if __has_builtin(__builtin_amdgcn_s_wait_asynccnt)
    __builtin_amdgcn_s_wait_asynccnt(N);
#else
    asm volatile("s_wait_asynccnt 0x0" ::: "memory");
#endif
}

// ---------------------------------------------------------------------------
// Generic bf16 WMMA GEMM:  C[M,N] (+= over NPAIR pairs)  A_p[M,K] * B_p[N,K]^T
// B is stored n-major ("B-transposed"): B[n*K + k].
// A_F32: A is fp32 in global (converted to bf16 on the way into LDS).
// OUT_BF16: store C as bf16 (else fp32). bias (len N) may be null.
// Double-buffered LDS; async global->LDS (ASYNCcnt pipeline) when available.
// ---------------------------------------------------------------------------
template <bool A_F32, bool OUT_BF16, int NPAIR>
__global__ __launch_bounds__(256) void gemm_bt_wmma(
    const void* __restrict__ Av0, const __bf16* __restrict__ B0,
    const void* __restrict__ Av1, const __bf16* __restrict__ B1,
    const float* __restrict__ bias, void* __restrict__ Cv,
    int M, int N, int K, int lda0, int lda1, int ldc)
{
    __shared__ __bf16 As[2][NPAIR][BLK_M * LDSS];
    __shared__ __bf16 Bs[2][NPAIR][BLK_N * LDSS];

    const int tid  = threadIdx.x;
    const int lane = tid & 31;
    const int wave = tid >> 5;          // 0..7
    const int wm   = wave & 3;          // 4 M-groups of 32 rows
    const int wn   = wave >> 2;         // 2 N-groups of 64 cols
    const int bm   = blockIdx.x * BLK_M;
    const int bn   = blockIdx.y * BLK_N;

    v8f zero = {};
    v8f acc[2][4];
#pragma unroll
    for (int mi = 0; mi < 2; ++mi)
#pragma unroll
        for (int ni = 0; ni < 4; ++ni) acc[mi][ni] = zero;

    // async b128 ops issued per thread per tile (A skipped when f32-converted)
    constexpr int TILE_OPS = (A_F32 ? 0 : 2 * NPAIR) + 2 * NPAIR;

    auto load_tile = [&](int buf, int kb) {
#pragma unroll
        for (int p = 0; p < NPAIR; ++p) {
            const void*   Ap  = p ? Av1 : Av0;
            const __bf16* Bp  = p ? B1  : B0;
            const int     lda = p ? lda1 : lda0;
#pragma unroll
            for (int c = 0; c < 2; ++c) {
                int idx = tid + c * 256;        // 0..511 chunk id
                int m   = idx >> 2;             // row in tile
                int kk  = (idx & 3) << 3;       // col base (0,8,16,24)
                __bf16* dstA = &As[buf][p][m * LDSS + kk];
                if constexpr (A_F32) {
                    const float* s = (const float*)Ap + (size_t)(bm + m) * lda + kb + kk;
#pragma unroll
                    for (int i = 0; i < 8; ++i) dstA[i] = (__bf16)s[i];
                } else {
                    const __bf16* s = (const __bf16*)Ap + (size_t)(bm + m) * lda + kb + kk;
#if HAVE_ASYNC_LDS
                    async_copy_b128(s, dstA);
#else
                    *(v8bf*)dstA = *(const v8bf*)s;
#endif
                }
                const __bf16* sB   = Bp + (size_t)(bn + m) * K + kb + kk;
                __bf16*       dstB = &Bs[buf][p][m * LDSS + kk];
#if HAVE_ASYNC_LDS
                async_copy_b128(sB, dstB);
#else
                *(v8bf*)dstB = *(const v8bf*)sB;
                if (kb + BLK_K < K) __builtin_prefetch(sB + BLK_K, 0, 1); // global_prefetch_b8
#endif
            }
        }
    };

    load_tile(0, 0);

    int buf = 0;
    for (int kb = 0; kb < K; kb += BLK_K, buf ^= 1) {
        const bool has_next = (kb + BLK_K) < K;
        if (has_next) load_tile(buf ^ 1, kb + BLK_K);   // prefetch next tile (in-flight)
#if HAVE_ASYNC_LDS
        // ASYNCcnt completes in order: <=TILE_OPS means current tile's copies landed
        if (has_next) wait_asynccnt<TILE_OPS>(); else wait_asynccnt<0>();
#endif
        __syncthreads();                                 // tile `buf` visible to all waves

        // ---- WMMA compute: fragments straight from LDS per ISA lane layout ----
        const int mr = lane & 15;
#pragma unroll
        for (int p = 0; p < NPAIR; ++p) {
            v16bf af[2];
            const int ka = (lane < 16) ? 0 : 8;          // A 16x32: K {ka..ka+7, ka+16..ka+23}
#pragma unroll
            for (int mi = 0; mi < 2; ++mi) {
                const __bf16* base = &As[buf][p][(wm * 32 + mi * 16 + mr) * LDSS + ka];
                v8bf lo = *(const v8bf*)base;
                v8bf hi = *(const v8bf*)(base + 16);
#pragma unroll
                for (int i = 0; i < 8; ++i) { af[mi][i] = lo[i]; af[mi][8 + i] = hi[i]; }
            }
            const int kbq = (lane < 16) ? 0 : 8;         // B 32x16: K {kbq..kbq+7, kbq+16..kbq+23}
#pragma unroll
            for (int ni = 0; ni < 4; ++ni) {
                const __bf16* base = &Bs[buf][p][(wn * 64 + ni * 16 + mr) * LDSS + kbq];
                v8bf lo = *(const v8bf*)base;
                v8bf hi = *(const v8bf*)(base + 16);
                v16bf bfr;
#pragma unroll
                for (int i = 0; i < 8; ++i) { bfr[i] = lo[i]; bfr[8 + i] = hi[i]; }
#pragma unroll
                for (int mi = 0; mi < 2; ++mi)
                    acc[mi][ni] = __builtin_amdgcn_wmma_f32_16x16x32_bf16(
                        false, af[mi], false, bfr, (short)0, acc[mi][ni], false, false);
            }
        }
        __syncthreads();                                 // all reads of `buf` done
    }

    // ---- store: C/D layout — VGPR r: lanes 0-15 -> M=r, lanes 16-31 -> M=8+r ----
#pragma unroll
    for (int mi = 0; mi < 2; ++mi) {
#pragma unroll
        for (int ni = 0; ni < 4; ++ni) {
            int n  = bn + wn * 64 + ni * 16 + (lane & 15);
            int m0 = bm + wm * 32 + mi * 16 + ((lane < 16) ? 0 : 8);
            float bv = bias ? bias[n] : 0.0f;
#pragma unroll
            for (int r = 0; r < 8; ++r) {
                float v = acc[mi][ni][r] + bv;
                if constexpr (OUT_BF16)
                    ((__bf16*)Cv)[(size_t)(m0 + r) * ldc + n] = (__bf16)v;
                else
                    ((float*)Cv)[(size_t)(m0 + r) * ldc + n] = v;
            }
        }
    }
}

// ---------------------------------------------------------------------------
// LN-LSTM elementwise cell: one block per batch row. pre: (512 x 4096),
// columns [0,1024)=i, [1024,2048)=f, [2048,3072)=o, [3072,4096)=c~.
// c state updated in place. Emits f32 h/c and bf16 h for next GEMM.
// ---------------------------------------------------------------------------
__global__ __launch_bounds__(256) void lstm_cell_ln(
    const float* __restrict__ pre, float* __restrict__ c_state,
    float* __restrict__ h_out, __bf16* __restrict__ h_bf)
{
    const int b   = blockIdx.x;
    const int tid = threadIdx.x;
    const float* row = pre + (size_t)b * 4096;
    float* crow = c_state + (size_t)b * 1024;

    float craw[4], og[4];
    float s = 0.f, s2 = 0.f;
#pragma unroll
    for (int j = 0; j < 4; ++j) {
        int h = tid + j * 256;
        float ig = 1.f / (1.f + __expf(-row[h]));
        float fg = 1.f / (1.f + __expf(-row[1024 + h]));
        float oo = 1.f / (1.f + __expf(-row[2048 + h]));
        float ct = tanhf(row[3072 + h]);
        float cr = fg * crow[h] + ig * ct;
        craw[j] = cr; og[j] = oo;
        s += cr; s2 += cr * cr;
    }
    __shared__ float rs[256], rq[256];
    rs[tid] = s; rq[tid] = s2;
    __syncthreads();
    for (int off = 128; off > 0; off >>= 1) {
        if (tid < off) { rs[tid] += rs[tid + off]; rq[tid] += rq[tid + off]; }
        __syncthreads();
    }
    float mean = rs[0] * (1.0f / 1024.0f);
    float var  = rq[0] * (1.0f / 1024.0f) - mean * mean;
    float rstd = rsqrtf(var + 1e-5f);
#pragma unroll
    for (int j = 0; j < 4; ++j) {
        int h = tid + j * 256;
        float cn = (craw[j] - mean) * rstd;
        float hv = og[j] * tanhf(cn);
        crow[h] = cn;
        h_out[(size_t)b * 1024 + h] = hv;
        h_bf [(size_t)b * 1024 + h] = (__bf16)hv;
    }
}

// ---------------------------------------------------------------------------
__global__ void f32_to_bf16_k(const float* __restrict__ s, __bf16* __restrict__ d, int n) {
    int i = blockIdx.x * 256 + threadIdx.x;
    if (i < n) d[i] = (__bf16)s[i];
}

// wh (4,1024,1024)[g][h][k2] -> Bt (4096 x 1024): dst[(g*1024+k2)*1024 + h]
__global__ void pack_wh_k(const float* __restrict__ s, __bf16* __restrict__ d) {
    int i = blockIdx.x * 256 + threadIdx.x;   // over 4096*1024
    if (i < 4096 * 1024) {
        int k  = i & 1023;       // = h
        int nn = i >> 10;        // = g*1024 + k2
        int g  = nn >> 10;
        int k2 = nn & 1023;
        d[i] = (__bf16)s[((size_t)g * 1024 + k) * 1024 + k2];
    }
}

__global__ void init_state_k(float* h0, float* c0, float* h1, float* c1,
                             __bf16* h0b, __bf16* h1b, int n) {
    int i = blockIdx.x * 256 + threadIdx.x;
    if (i < n) {
        h0[i] = 0.f; c0[i] = 0.f; h1[i] = 0.f; c1[i] = 0.f;
        h0b[i] = (__bf16)0.f; h1b[i] = (__bf16)0.f;
    }
}

__global__ void gather_out_k(const float* h0, const float* h1,
                             const float* c0, const float* c1,
                             float* out, int n) {
    int i = blockIdx.x * 256 + threadIdx.x;
    if (i < n) {
        out[(size_t)n * 1 + i] = h0[i];   // after logit block (n == 512*1024)
        out[(size_t)n * 2 + i] = h1[i];
        out[(size_t)n * 3 + i] = c0[i];
        out[(size_t)n * 4 + i] = c1[i];
    }
}

// ---------------------------------------------------------------------------
extern "C" void kernel_launch(void* const* d_in, const int* in_sizes, int n_in,
                              void* d_out, int out_size, void* d_ws, size_t ws_size,
                              hipStream_t stream) {
    (void)in_sizes; (void)n_in; (void)out_size; (void)ws_size;

    const float* obs   = (const float*)d_in[0];
    const float* fc1_w = (const float*)d_in[1];
    const float* fc1_b = (const float*)d_in[2];
    const float* wx1   = (const float*)d_in[3];
    const float* wh1   = (const float*)d_in[4];
    const float* b1    = (const float*)d_in[5];
    const float* wx2   = (const float*)d_in[6];
    const float* wh2   = (const float*)d_in[7];
    const float* b2    = (const float*)d_in[8];
    const float* fc2_w = (const float*)d_in[9];
    const float* fc2_b = (const float*)d_in[10];
    float* out = (float*)d_out;

    const int BS = 512, SEQ = 64, OBS = 512, H = 1024;
    const int M1 = BS * SEQ;          // 32768
    const int BH = BS * H;            // 524288

    char* w = (char*)d_ws;
    size_t off = 0;
    auto alloc = [&](size_t bytes) -> void* {
        void* p = w + off;
        off = (off + bytes + 255) & ~(size_t)255;
        return p;
    };
    __bf16* x_bf    = (__bf16*)alloc((size_t)M1 * H * 2);      // 64 MB
    __bf16* fc1w_bf = (__bf16*)alloc((size_t)H * OBS * 2);
    __bf16* wx1_bf  = (__bf16*)alloc((size_t)4 * H * H * 2);
    __bf16* wh1p_bf = (__bf16*)alloc((size_t)4 * H * H * 2);
    __bf16* wx2_bf  = (__bf16*)alloc((size_t)4 * H * H * 2);
    __bf16* wh2p_bf = (__bf16*)alloc((size_t)4 * H * H * 2);
    __bf16* fc2w_bf = (__bf16*)alloc((size_t)H * H * 2);
    float*  pre     = (float*) alloc((size_t)BS * 4 * H * 4);  // 8 MB
    float*  h0      = (float*) alloc((size_t)BH * 4);
    float*  c0      = (float*) alloc((size_t)BH * 4);
    float*  h1      = (float*) alloc((size_t)BH * 4);
    float*  c1      = (float*) alloc((size_t)BH * 4);
    __bf16* h0_bf   = (__bf16*)alloc((size_t)BH * 2);
    __bf16* h1_bf   = (__bf16*)alloc((size_t)BH * 2);

    auto g1 = [](int n) { return dim3((n + 255) / 256); };

    // ---- one-time weight conversion / packing ----
    f32_to_bf16_k<<<g1(H * OBS),   256, 0, stream>>>(fc1_w, fc1w_bf, H * OBS);
    f32_to_bf16_k<<<g1(4 * H * H), 256, 0, stream>>>(wx1, wx1_bf, 4 * H * H);
    f32_to_bf16_k<<<g1(4 * H * H), 256, 0, stream>>>(wx2, wx2_bf, 4 * H * H);
    f32_to_bf16_k<<<g1(H * H),     256, 0, stream>>>(fc2_w, fc2w_bf, H * H);
    pack_wh_k<<<g1(4 * H * H), 256, 0, stream>>>(wh1, wh1p_bf);
    pack_wh_k<<<g1(4 * H * H), 256, 0, stream>>>(wh2, wh2p_bf);
    init_state_k<<<g1(BH), 256, 0, stream>>>(h0, c0, h1, c1, h0_bf, h1_bf, BH);

    // ---- fc1 over all timesteps: x = obs @ fc1_w^T + b  (bf16 out) ----
    gemm_bt_wmma<true, true, 1><<<dim3(M1 / BLK_M, H / BLK_N), 256, 0, stream>>>(
        obs, fc1w_bf, nullptr, nullptr, fc1_b, x_bf,
        M1, H, OBS, OBS, 0, H);

    // ---- recurrent scan: 64 steps x 2 layers ----
    for (int t = 0; t < SEQ; ++t) {
        // layer 1: pre = x_t @ Wx1^T + h0 @ Wh1 + b1
        gemm_bt_wmma<false, false, 2><<<dim3(BS / BLK_M, 4 * H / BLK_N), 256, 0, stream>>>(
            x_bf + (size_t)t * H, wx1_bf, h0_bf, wh1p_bf, b1, pre,
            BS, 4 * H, H, SEQ * H, H, 4 * H);
        lstm_cell_ln<<<BS, 256, 0, stream>>>(pre, c0, h0, h0_bf);

        // layer 2: pre = h0 @ Wx2^T + h1 @ Wh2 + b2
        gemm_bt_wmma<false, false, 2><<<dim3(BS / BLK_M, 4 * H / BLK_N), 256, 0, stream>>>(
            h0_bf, wx2_bf, h1_bf, wh2p_bf, b2, pre,
            BS, 4 * H, H, H, H, 4 * H);
        lstm_cell_ln<<<BS, 256, 0, stream>>>(pre, c1, h1, h1_bf);
    }

    // ---- fc2: logit = h1 @ fc2_w^T + b  (f32 into d_out) ----
    gemm_bt_wmma<false, false, 1><<<dim3(BS / BLK_M, H / BLK_N), 256, 0, stream>>>(
        h1_bf, fc2w_bf, nullptr, nullptr, fc2_b, out,
        BS, H, H, H, 0, H);

    // ---- h_n, c_n ----
    gather_out_k<<<g1(BH), 256, 0, stream>>>(h0, h1, c0, c1, out, BH);
}
